// NonOverlappingLocallyConnected1d_new_76081050681911
// MI455X (gfx1250) — matrix operand order repaired and checked by reading, not compiled, not run on hardware
//
#include <hip/hip_runtime.h>

typedef __attribute__((ext_vector_type(16))) __bf16       v16bf;
typedef __attribute__((ext_vector_type(2)))  __bf16       v2bf;
typedef __attribute__((ext_vector_type(8)))  float        v8f;
typedef __attribute__((ext_vector_type(4)))  unsigned int v4u;

#define M_TOT 1024
#define N_TOT 512
#define K_TOT 32768   /* G*CIN = 64*512 */
#define TM 128
#define TN 64
#define TK 32
#define LDA 40        /* bf16 elems per A row in LDS (32 + 8 pad) */
#define LDB 40        /* bf16 elems per B column in LDS */
#define SPLITK 4

__device__ __forceinline__ unsigned int pkbf(float a, float b) {
  // Build a packed bf16 pair directly -> should lower to one v_cvt_pk_bf16_f32.
  v2bf v;
  v.x = (__bf16)a;
  v.y = (__bf16)b;
  return __builtin_bit_cast(unsigned int, v);
}

__global__ __launch_bounds__(256)
void glc1d_gemm_bf16(const float* __restrict__ X,
                     const float* __restrict__ W,
                     float* __restrict__ Out,
                     float* __restrict__ Ws,
                     int kSpan, float outScale)
{
  __shared__ __align__(16) unsigned short lA[2][TM * LDA];  // 2 x 10.0 KB
  __shared__ __align__(16) unsigned short lB[2][TN * LDB];  // 2 x  5.0 KB

  const int tid  = threadIdx.x;
  const int lane = tid & 31;
  const int wid  = tid >> 5;
  const int wm   = wid & 3;    // wave's 32-row strip (M)
  const int wn   = wid >> 2;   // wave's 32-col strip (N)
  const int h    = lane >> 4;  // K-half selector within fragments
  const int lm   = lane & 15;

  const int blockM = blockIdx.y * TM;
  const int blockN = blockIdx.x * TN;
  const int kz     = blockIdx.z;
  const int k0     = kz * kSpan;
  const int kEnd   = k0 + kSpan;

  float* __restrict__ dst =
      (kz == 0) ? Out : (Ws + (size_t)(kz - 1) * (M_TOT * N_TOT));

  // A staging: thread -> (row group ar0, k-quad aq)
  const int aq  = tid & 7;     // floats 4*aq .. 4*aq+3 of the 32-wide k tile
  const int ar0 = tid >> 3;    // 0..31 ; rows ar0 + 32*i
  // B staging: thread -> one tile column, 8 consecutive k values
  const int bn  = tid & 63;    // 0..63 column within tile
  const int bks = tid >> 6;    // 0..3 -> k segment of 8

  float4 aReg[4];
  float  bReg[8];

  const float* aBase = X + (size_t)blockM * K_TOT;
  const float* wBase = W + blockN;

  // ---- prologue: stage first k-tile, commit to LDS buffer 0 ----
  {
    const float* ap = aBase + k0 + aq * 4;
#pragma unroll
    for (int i = 0; i < 4; ++i)
      aReg[i] = *reinterpret_cast<const float4*>(ap + (size_t)(ar0 + 32 * i) * K_TOT);
    const float* wp = wBase + (size_t)(k0 + bks * 8) * N_TOT + bn;
#pragma unroll
    for (int j = 0; j < 8; ++j)
      bReg[j] = wp[(size_t)j * N_TOT];
  }
#pragma unroll
  for (int i = 0; i < 4; ++i) {
    *reinterpret_cast<uint2*>(&lA[0][(ar0 + 32 * i) * LDA + aq * 4]) =
        make_uint2(pkbf(aReg[i].x, aReg[i].y), pkbf(aReg[i].z, aReg[i].w));
  }
  {
    v4u bp;
    bp.x = pkbf(bReg[0], bReg[1]);
    bp.y = pkbf(bReg[2], bReg[3]);
    bp.z = pkbf(bReg[4], bReg[5]);
    bp.w = pkbf(bReg[6], bReg[7]);
    *reinterpret_cast<v4u*>(&lB[0][bn * LDB + bks * 8]) = bp;
  }
  __syncthreads();

  v8f acc[2][2] = {};
  int p = 0;

  for (int kt = k0; kt < kEnd; kt += TK) {
    const bool hasNext = (kt + TK < kEnd);

    // issue next k-tile's global loads early; consumed after the WMMAs
    if (hasNext) {
      const float* ap = aBase + (kt + TK) + aq * 4;
#pragma unroll
      for (int i = 0; i < 4; ++i)
        aReg[i] = *reinterpret_cast<const float4*>(ap + (size_t)(ar0 + 32 * i) * K_TOT);
      const float* wp = wBase + (size_t)(kt + TK + bks * 8) * N_TOT + bn;
#pragma unroll
      for (int j = 0; j < 8; ++j)
        bReg[j] = wp[(size_t)j * N_TOT];
    }

    // WMMA fragments from LDS buffer p (two contiguous 16B runs per lane,
    // matching the CDNA5 16-bit A(16x32) / B(32x16) VGPR layouts).
    const unsigned short* __restrict__ cA = lA[p];
    const unsigned short* __restrict__ cB = lB[p];
    v16bf aF[2], bF[2];
#pragma unroll
    for (int mi = 0; mi < 2; ++mi) {
      int row = wm * 32 + mi * 16 + lm;
      v4u lo = *reinterpret_cast<const v4u*>(&cA[row * LDA + h * 8]);       // K = h*8 .. +7
      v4u hi = *reinterpret_cast<const v4u*>(&cA[row * LDA + 16 + h * 8]);  // K = 16+h*8 ..
      aF[mi] = __builtin_bit_cast(v16bf,
                 __builtin_shufflevector(lo, hi, 0, 1, 2, 3, 4, 5, 6, 7));
    }
#pragma unroll
    for (int ni = 0; ni < 2; ++ni) {
      int col = wn * 32 + ni * 16 + lm;
      v4u lo = *reinterpret_cast<const v4u*>(&cB[col * LDB + h * 8]);       // K = h*8 .. +7
      v4u hi = *reinterpret_cast<const v4u*>(&cB[col * LDB + 16 + h * 8]);  // K = 16+h*8 ..
      bF[ni] = __builtin_bit_cast(v16bf,
                 __builtin_shufflevector(lo, hi, 0, 1, 2, 3, 4, 5, 6, 7));
    }

#pragma unroll
    for (int mi = 0; mi < 2; ++mi)
#pragma unroll
      for (int ni = 0; ni < 2; ++ni)
        acc[mi][ni] = __builtin_amdgcn_wmma_f32_16x16x32_bf16(
            false, aF[mi], false, bF[ni], (short)0, acc[mi][ni], false, false);

    // commit next tile to the other LDS buffer; single barrier per iteration
    if (hasNext) {
      unsigned short* __restrict__ nA = lA[p ^ 1];
      unsigned short* __restrict__ nB = lB[p ^ 1];
#pragma unroll
      for (int i = 0; i < 4; ++i) {
        *reinterpret_cast<uint2*>(&nA[(ar0 + 32 * i) * LDA + aq * 4]) =
            make_uint2(pkbf(aReg[i].x, aReg[i].y), pkbf(aReg[i].z, aReg[i].w));
      }
      v4u bp;
      bp.x = pkbf(bReg[0], bReg[1]);
      bp.y = pkbf(bReg[2], bReg[3]);
      bp.z = pkbf(bReg[4], bReg[5]);
      bp.w = pkbf(bReg[6], bReg[7]);
      *reinterpret_cast<v4u*>(&nB[bn * LDB + bks * 8]) = bp;
      __syncthreads();
      p ^= 1;
    }
  }

  // epilogue: optional scale, store fp32 partial/result (no atomics)
#pragma unroll
  for (int mi = 0; mi < 2; ++mi) {
#pragma unroll
    for (int ni = 0; ni < 2; ++ni) {
      int col = blockN + wn * 32 + ni * 16 + lm;
#pragma unroll
      for (int j = 0; j < 8; ++j) {
        int row = blockM + wm * 32 + mi * 16 + h * 8 + j;
        dst[(size_t)row * N_TOT + col] = acc[mi][ni][j] * outScale;
      }
    }
  }
}

// out = (out + ws0 + ws1 + ws2) * scale, vectorized float4
__global__ __launch_bounds__(256)
void glc1d_reduce(float* __restrict__ Out, const float* __restrict__ Ws,
                  float scale)
{
  const size_t i = ((size_t)blockIdx.x * 256 + threadIdx.x) * 4;
  float4 o = *reinterpret_cast<float4*>(Out + i);
  float4 a = *reinterpret_cast<const float4*>(Ws + i);
  float4 b = *reinterpret_cast<const float4*>(Ws + i + (size_t)M_TOT * N_TOT);
  float4 c = *reinterpret_cast<const float4*>(Ws + i + 2 * (size_t)M_TOT * N_TOT);
  o.x = (o.x + a.x + b.x + c.x) * scale;
  o.y = (o.y + a.y + b.y + c.y) * scale;
  o.z = (o.z + a.z + b.z + c.z) * scale;
  o.w = (o.w + a.w + b.w + c.w) * scale;
  *reinterpret_cast<float4*>(Out + i) = o;
}

extern "C" void kernel_launch(void* const* d_in, const int* in_sizes, int n_in,
                              void* d_out, int out_size, void* d_ws, size_t ws_size,
                              hipStream_t stream) {
  (void)in_sizes; (void)n_in; (void)out_size;
  const float* x = (const float*)d_in[0];   // [1024, 64, 512] fp32 == A[1024, 32768]
  const float* w = (const float*)d_in[1];   // [64, 512, 512]  fp32 == W[32768, 512]
  float* out = (float*)d_out;               // [1024, 512] fp32
  const float scale = 0.04419417382415922f; // 1/sqrt(512)

  const size_t partialBytes = (size_t)(SPLITK - 1) * M_TOT * N_TOT * sizeof(float);
  if (ws_size >= partialBytes) {
    // split-K=4: 256 workgroups, deterministic two-kernel reduction
    dim3 grid(N_TOT / TN, M_TOT / TM, SPLITK);
    glc1d_gemm_bf16<<<grid, 256, 0, stream>>>(x, w, out, (float*)d_ws,
                                              K_TOT / SPLITK, 1.0f);
    const int nvec = (M_TOT * N_TOT) / 4;   // 131072 float4 elements
    glc1d_reduce<<<nvec / 256, 256, 0, stream>>>(out, (const float*)d_ws, scale);
  } else {
    // fallback: single pass, scale in-kernel
    dim3 grid(N_TOT / TN, M_TOT / TM, 1);
    glc1d_gemm_bf16<<<grid, 256, 0, stream>>>(x, w, out, nullptr, K_TOT, scale);
  }
}
